// LGCNModelBase_65644280152554
// MI455X (gfx1250) — compile-verified, matrix operand order checked by reading, not run on hardware
//
#include <hip/hip_runtime.h>

#define INTD 21
#define NTESTS 1538
#define NQUEST 9455
#define NTAGS  913
#define NUSERS 7442
#define GDIM   64
#define NTOK   (512 * 1024)

typedef __attribute__((ext_vector_type(16))) _Float16 v16h;
typedef __attribute__((ext_vector_type(8)))  float    v8f;

// padded row counts (multiples of 16) for projected tables in workspace
#define RP_INT  16
#define RP_TEST 1552
#define RP_Q    9456
#define RP_TAG  928

// ---------------- workspace layout (float offsets) ----------------
#define WS_WQ    0                        // 64x32 fused graph-q weight (f32)
#define WS_WT    2048
#define WS_WG    4096
#define WS_BTOT  6144                     // folded bias [32]
#define WS_TINT  6176                     // RP_INT  x32 projected tables (row-padded)
#define WS_TTEST (WS_TINT  + RP_INT  * 32)
#define WS_TQ    (WS_TTEST + RP_TEST * 32)
#define WS_TTAG  (WS_TQ    + RP_Q    * 32)
#define WS_TGQ   (WS_TTAG  + RP_TAG  * 32)
#define WS_TGT   (WS_TGQ   + RP_Q    * 32)
#define WS_TGG   (WS_TGT   + RP_TEST * 32)
#define WS_PACK  (WS_TGG   + RP_TAG  * 32) // packed f16 B fragments (20480 halfs)
// half-offsets inside pack region:
//   cat hi: seg*1024          cat lo: 4096 + seg*1024
//   grf hi: 8192 + g*2048     grf lo: 14336 + g*2048

// ============ 1) fold graph linears + biases through comb_w ============
__global__ void fold_weights_kernel(const float* __restrict__ glqw, const float* __restrict__ glqb,
                                    const float* __restrict__ gltw, const float* __restrict__ gltb,
                                    const float* __restrict__ glgw, const float* __restrict__ glgb,
                                    const float* __restrict__ comb_w, const float* __restrict__ comb_b,
                                    float* __restrict__ ws) {
    const float* glw[3] = {glqw, gltw, glgw};
    const float* glb[3] = {glqb, gltb, glgb};
    float* W[3] = {ws + WS_WQ, ws + WS_WT, ws + WS_WG};
    for (int tid = threadIdx.x; tid < 3 * 2048 + 32; tid += blockDim.x) {
        if (tid < 3 * 2048) {
            int mat = tid / 2048, rem = tid % 2048;
            int d = rem / 32, c = rem % 32;
            const float* cseg = comb_w + (4 + mat) * INTD * 32;  // segments 4,5,6
            float acc = 0.f;
            for (int k = 0; k < INTD; ++k)
                acc += glw[mat][d * INTD + k] * cseg[k * 32 + c];
            W[mat][d * 32 + c] = acc;
        } else {
            int c = tid - 3 * 2048;
            float acc = comb_b[c];
            for (int mat = 0; mat < 3; ++mat) {
                const float* cseg = comb_w + (4 + mat) * INTD * 32;
                for (int k = 0; k < INTD; ++k)
                    acc += glb[mat][k] * cseg[k * 32 + c];
            }
            ws[WS_BTOT + c] = acc;
        }
    }
}

// ============ 2) pack all 7 B matrices into WMMA fragment layout (hi/lo f16) ============
__global__ void pack_b_kernel(const float* __restrict__ comb_w, float* __restrict__ ws) {
    int wave = (blockIdx.x * blockDim.x + threadIdx.x) >> 5;
    int lane = threadIdx.x & 31;
    int hi = lane >> 4, mn = lane & 15;
    _Float16* wh = (_Float16*)(ws + WS_PACK);
    const float* W;
    int K, s, colTile;
    _Float16 *dst_hi, *dst_lo;
    if (wave < 8) {                      // 4 categorical segs x 2 colTiles, 1 k-step
        int seg = wave >> 1; colTile = wave & 1; s = 0; K = INTD;
        W = comb_w + seg * INTD * 32;
        dst_hi = wh + seg * 1024;
        dst_lo = wh + 4096 + seg * 1024;
    } else if (wave < 20) {              // 3 graph weights x 2 steps x 2 colTiles
        int w2 = wave - 8;
        int g = w2 >> 2, r = w2 & 3;
        s = r >> 1; colTile = r & 1; K = GDIM;
        W = ws + WS_WQ + g * 2048;
        dst_hi = wh + 8192  + g * 2048;
        dst_lo = wh + 14336 + g * 2048;
    } else return;
    v16h fh, fl;
    #pragma unroll
    for (int h = 0; h < 16; ++h) {
        // ISA 16-bit B 32x16 VGPR layout (K across halfs, N = lane&15)
        int kl = (h < 8) ? (8 * hi + h) : (16 + 8 * hi + (h - 8));
        int k  = (s << 5) + kl;
        float v = (k < K) ? W[k * 32 + colTile * 16 + mn] : 0.f;
        _Float16 vh = (_Float16)v;
        fh[h] = vh;
        fl[h] = (_Float16)(v - (float)vh);
    }
    int fidx = (s * 2 + colTile) * 32 + lane;
    *(v16h*)(dst_hi + fidx * 16) = fh;
    *(v16h*)(dst_lo + fidx * 16) = fl;
}

// ============ 3) table projection: out[Rpad,32] = in[rowOffset+r, K] @ W[K,32] ============
// One wave per 16x16 tile; split-f16 (hi/lo) -> 3 WMMAs per k-step for ~f32 accuracy.
// Output table is row-padded to a multiple of 16 -> unguarded straight-line stores.
template <int K>
__global__ void table_project_wmma(const float* __restrict__ in, int rowOffset,
                                   const _Float16* __restrict__ packB_hi,
                                   const _Float16* __restrict__ packB_lo,
                                   int R, float* __restrict__ out) {
    constexpr int KSTEPS = (K + 31) / 32;
    int wave = (blockIdx.x * blockDim.x + threadIdx.x) >> 5;
    int lane = threadIdx.x & 31;
    int rowTile = wave >> 1, colTile = wave & 1;
    int rowBase = rowTile * 16;
    if (rowBase >= R) return;            // wave-uniform -> EXEC all-ones at WMMA
    int hi = lane >> 4, mn = lane & 15;
    int row = rowBase + mn;
    row = (row < R) ? row : (R - 1);     // clamp loads: OOB A rows feed only pad rows

    v8f acc = {};
    #pragma unroll
    for (int s = 0; s < KSTEPS; ++s) {
        v16h ah, al;
        if constexpr (K == GDIM) {
            // two contiguous 8-float runs per lane, 32B aligned -> 4x b128 loads
            const float* rp = in + (long)(rowOffset + row) * GDIM + (s << 5) + 8 * hi;
            float4 f0 = *(const float4*)(rp);
            float4 f1 = *(const float4*)(rp + 4);
            float4 f2 = *(const float4*)(rp + 16);
            float4 f3 = *(const float4*)(rp + 20);
            float av[16] = {f0.x, f0.y, f0.z, f0.w, f1.x, f1.y, f1.z, f1.w,
                            f2.x, f2.y, f2.z, f2.w, f3.x, f3.y, f3.z, f3.w};
            #pragma unroll
            for (int h = 0; h < 16; ++h) {
                _Float16 vh = (_Float16)av[h];
                ah[h] = vh;
                al[h] = (_Float16)(av[h] - (float)vh);
            }
        } else {
            // K=21: scalar loads, branchless clamp + select (no exec-mask dance)
            long base   = (long)(rowOffset + row) * K;
            long maxIdx = (long)(rowOffset + R) * K - 1;
            #pragma unroll
            for (int h = 0; h < 16; ++h) {
                int  kl  = (h < 8) ? (8 * hi + h) : (16 + 8 * hi + (h - 8));
                long idx = base + kl;
                idx = (idx < maxIdx) ? idx : maxIdx;
                float v = in[idx];
                v = (kl < K) ? v : 0.f;
                _Float16 vh = (_Float16)v;
                ah[h] = vh;
                al[h] = (_Float16)(v - (float)vh);
            }
        }
        int fidx = (s * 2 + colTile) * 32 + lane;
        v16h bh = *(const v16h*)(packB_hi + fidx * 16);
        v16h bl = *(const v16h*)(packB_lo + fidx * 16);
        acc = __builtin_amdgcn_wmma_f32_16x16x32_f16(false, al, false, bh, (short)0, acc, false, false);
        acc = __builtin_amdgcn_wmma_f32_16x16x32_f16(false, ah, false, bl, (short)0, acc, false, false);
        acc = __builtin_amdgcn_wmma_f32_16x16x32_f16(false, ah, false, bh, (short)0, acc, false, false);
    }
    // D layout: VGPR v -> row rowBase + v + 8*(lane>>4), col cb + (lane&15).
    // Table is padded to 16-row multiple: store all 8 unconditionally.
    float* op = out + (long)(rowBase + 8 * hi) * 32 + colTile * 16 + mn;
    #pragma unroll
    for (int v = 0; v < 8; ++v)
        op[v * 32] = acc[v];
}

// ============ 4) fused gather + sum + dual LayerNorm, 8 threads per token ============
__global__ __launch_bounds__(256) void fused_embed_ln(
    const int* __restrict__ test, const int* __restrict__ question,
    const int* __restrict__ tag,  const int* __restrict__ interaction,
    const float* __restrict__ elapsed, const float* __restrict__ ws,
    const float* __restrict__ comb_ln_g, const float* __restrict__ comb_ln_b,
    const float* __restrict__ cont_w, const float* __restrict__ cont_b,
    const float* __restrict__ cont_ln_g, const float* __restrict__ cont_ln_b,
    float* __restrict__ out) {
    int gtid = blockIdx.x * 256 + threadIdx.x;
    int tok  = gtid >> 3;
    int c0   = (gtid & 7) * 4;

    int it = test[tok], iq = question[tok], ig = tag[tok], ii = interaction[tok];
    float el = elapsed[tok];

    const float* Tint  = ws + WS_TINT;
    const float* Ttest = ws + WS_TTEST;
    const float* Tq    = ws + WS_TQ;
    const float* Ttag  = ws + WS_TTAG;
    const float* Tgq   = ws + WS_TGQ;
    const float* Tgt   = ws + WS_TGT;
    const float* Tgg   = ws + WS_TGG;
    const float* btot  = ws + WS_BTOT;

    #define LD4(p) (*(const float4*)(p))
    float4 a0 = LD4(Tint  + ii * 32 + c0);
    float4 a1 = LD4(Ttest + it * 32 + c0);
    float4 a2 = LD4(Tq    + iq * 32 + c0);
    float4 a3 = LD4(Ttag  + ig * 32 + c0);
    float4 a4 = LD4(Tgq   + iq * 32 + c0);
    float4 a5 = LD4(Tgt   + it * 32 + c0);
    float4 a6 = LD4(Tgg   + ig * 32 + c0);
    float4 ab = LD4(btot + c0);

    float x[4];
    x[0] = a0.x + a1.x + a2.x + a3.x + a4.x + a5.x + a6.x + ab.x;
    x[1] = a0.y + a1.y + a2.y + a3.y + a4.y + a5.y + a6.y + ab.y;
    x[2] = a0.z + a1.z + a2.z + a3.z + a4.z + a5.z + a6.z + ab.z;
    x[3] = a0.w + a1.w + a2.w + a3.w + a4.w + a5.w + a6.w + ab.w;

    // LayerNorm over 32 channels: width-8 xor-shuffle tree inside wave32
    float s  = x[0] + x[1] + x[2] + x[3];
    float sq = x[0]*x[0] + x[1]*x[1] + x[2]*x[2] + x[3]*x[3];
    s  += __shfl_xor(s, 4, 8);  sq += __shfl_xor(sq, 4, 8);
    s  += __shfl_xor(s, 2, 8);  sq += __shfl_xor(sq, 2, 8);
    s  += __shfl_xor(s, 1, 8);  sq += __shfl_xor(sq, 1, 8);
    float mean = s * 0.03125f;
    float inv  = rsqrtf(sq * 0.03125f - mean * mean + 1e-5f);

    float4 g  = LD4(comb_ln_g + c0);
    float4 bb = LD4(comb_ln_b + c0);
    float4 o;
    o.x = (x[0] - mean) * inv * g.x + bb.x;
    o.y = (x[1] - mean) * inv * g.y + bb.y;
    o.z = (x[2] - mean) * inv * g.z + bb.z;
    o.w = (x[3] - mean) * inv * g.w + bb.w;
    *(float4*)(out + (long)tok * 64 + c0) = o;

    // continuous path: elapsed * cont_w + cont_b, then LayerNorm(32)
    float4 cw = LD4(cont_w + c0);
    float4 cb = LD4(cont_b + c0);
    float y[4] = { el * cw.x + cb.x, el * cw.y + cb.y, el * cw.z + cb.z, el * cw.w + cb.w };
    float s2  = y[0] + y[1] + y[2] + y[3];
    float sq2 = y[0]*y[0] + y[1]*y[1] + y[2]*y[2] + y[3]*y[3];
    s2  += __shfl_xor(s2, 4, 8);  sq2 += __shfl_xor(sq2, 4, 8);
    s2  += __shfl_xor(s2, 2, 8);  sq2 += __shfl_xor(sq2, 2, 8);
    s2  += __shfl_xor(s2, 1, 8);  sq2 += __shfl_xor(sq2, 1, 8);
    float mean2 = s2 * 0.03125f;
    float inv2  = rsqrtf(sq2 * 0.03125f - mean2 * mean2 + 1e-5f);

    float4 g2  = LD4(cont_ln_g + c0);
    float4 bb2 = LD4(cont_ln_b + c0);
    float4 o2;
    o2.x = (y[0] - mean2) * inv2 * g2.x + bb2.x;
    o2.y = (y[1] - mean2) * inv2 * g2.y + bb2.y;
    o2.z = (y[2] - mean2) * inv2 * g2.z + bb2.z;
    o2.w = (y[3] - mean2) * inv2 * g2.w + bb2.w;
    *(float4*)(out + (long)tok * 64 + 32 + c0) = o2;
    #undef LD4
}

extern "C" void kernel_launch(void* const* d_in, const int* in_sizes, int n_in,
                              void* d_out, int out_size, void* d_ws, size_t ws_size,
                              hipStream_t stream) {
    (void)in_sizes; (void)n_in; (void)out_size; (void)ws_size;
    const int*   test        = (const int*)d_in[0];
    const int*   question    = (const int*)d_in[1];
    const int*   tag         = (const int*)d_in[2];
    const int*   interaction = (const int*)d_in[5];
    const float* elapsed     = (const float*)d_in[6];
    const float* emb_int     = (const float*)d_in[7];
    const float* emb_test    = (const float*)d_in[8];
    const float* emb_q       = (const float*)d_in[9];
    const float* emb_tag     = (const float*)d_in[10];
    const float* ge_q        = (const float*)d_in[11];
    const float* ge_t        = (const float*)d_in[12];
    const float* ge_g        = (const float*)d_in[13];
    const float* glqw        = (const float*)d_in[14];
    const float* glqb        = (const float*)d_in[15];
    const float* gltw        = (const float*)d_in[16];
    const float* gltb        = (const float*)d_in[17];
    const float* glgw        = (const float*)d_in[18];
    const float* glgb        = (const float*)d_in[19];
    const float* comb_w      = (const float*)d_in[20];
    const float* comb_b      = (const float*)d_in[21];
    const float* comb_ln_g   = (const float*)d_in[22];
    const float* comb_ln_b   = (const float*)d_in[23];
    const float* cont_w      = (const float*)d_in[24];
    const float* cont_b      = (const float*)d_in[25];
    const float* cont_ln_g   = (const float*)d_in[26];
    const float* cont_ln_b   = (const float*)d_in[27];
    float* ws  = (float*)d_ws;
    float* out = (float*)d_out;
    _Float16* wh = (_Float16*)(ws + WS_PACK);

    fold_weights_kernel<<<1, 256, 0, stream>>>(glqw, glqb, gltw, gltb, glgw, glgb,
                                               comb_w, comb_b, ws);
    pack_b_kernel<<<5, 128, 0, stream>>>(comb_w, ws);   // 20 fragment waves

    auto grid_for = [](int R) {
        int waves = ((R + 15) / 16) * 2;
        return (waves * 32 + 127) / 128;
    };
    // categorical tables @ comb_w segments 0..3 (K=21)
    table_project_wmma<INTD><<<grid_for(3), 128, 0, stream>>>(
        emb_int, 0, wh + 0 * 1024, wh + 4096 + 0 * 1024, 3, ws + WS_TINT);
    table_project_wmma<INTD><<<grid_for(NTESTS + 1), 128, 0, stream>>>(
        emb_test, 0, wh + 1 * 1024, wh + 4096 + 1 * 1024, NTESTS + 1, ws + WS_TTEST);
    table_project_wmma<INTD><<<grid_for(NQUEST + 1), 128, 0, stream>>>(
        emb_q, 0, wh + 2 * 1024, wh + 4096 + 2 * 1024, NQUEST + 1, ws + WS_TQ);
    table_project_wmma<INTD><<<grid_for(NTAGS + 1), 128, 0, stream>>>(
        emb_tag, 0, wh + 3 * 1024, wh + 4096 + 3 * 1024, NTAGS + 1, ws + WS_TTAG);
    // graph tables @ fused 64x32 weights (K=64), gathered at offset NUSERS-1
    table_project_wmma<GDIM><<<grid_for(NQUEST + 1), 128, 0, stream>>>(
        ge_q, NUSERS - 1, wh + 8192 + 0 * 2048, wh + 14336 + 0 * 2048, NQUEST + 1, ws + WS_TGQ);
    table_project_wmma<GDIM><<<grid_for(NTESTS + 1), 128, 0, stream>>>(
        ge_t, NUSERS - 1, wh + 8192 + 1 * 2048, wh + 14336 + 1 * 2048, NTESTS + 1, ws + WS_TGT);
    table_project_wmma<GDIM><<<grid_for(NTAGS + 1), 128, 0, stream>>>(
        ge_g, NUSERS - 1, wh + 8192 + 2 * 2048, wh + 14336 + 2 * 2048, NTAGS + 1, ws + WS_TGG);

    // main fused pass: 8 threads per token
    int blocks = (NTOK * 8) / 256;   // exact: 16384
    fused_embed_ln<<<blocks, 256, 0, stream>>>(test, question, tag, interaction, elapsed, ws,
                                               comb_ln_g, comb_ln_b, cont_w, cont_b,
                                               cont_ln_g, cont_ln_b, out);
}
